// GGNN_63548336111743
// MI455X (gfx1250) — matrix operand order repaired and checked by reading, not compiled, not run on hardware
//
#include <hip/hip_runtime.h>
#include <hip/hip_bf16.h>
#include <math.h>

typedef __bf16 bf16;
typedef __attribute__((ext_vector_type(16))) __bf16 v16bf;
typedef __attribute__((ext_vector_type(8)))  __bf16 v8bf;
typedef __attribute__((ext_vector_type(8)))  float  v8f;

#define DD     100
#define KP     128          // 100 padded to 128
#define NN     64
#define BBATCH 512
#define ROWS   (BBATCH * NN)     // 32768
#define NNODES 50000
#define NOUT   49999
#define NOUTP  50000             // ceil16(49999)
#define K3     300
#define K3P    320               // 300 padded to 320
#define XHW    320               // xh row width (Kpad of gates GEMM)
#define NFP    112               // ceil16(100)

// ---------------------------------------------------------------------------
// bf16 WMMA GEMM: C[M,N] = act(A[M,Kpad] * Bt^T + bias)
//   A : row-major [M, Kpad], lda = Kpad (mult of 8), zero-padded K region.
//   Bt: row-major [ceil16(N), Kpad] = B transposed, ldb mult of 8, padded.
//   Inner loop: 4 x b128 loads + 1 wmma, branch-free.
//   ctrans: store bf16 result transposed Cb[n*ldc + m] (one v8bf per lane).
//   act: 0 none, 1 sigmoid, 2 tanh.
// ---------------------------------------------------------------------------
__global__ __launch_bounds__(256)
void wmma_gemm(const bf16* __restrict__ A, int lda,
               const bf16* __restrict__ Bt, int ldb,
               const float* __restrict__ bias,
               float* __restrict__ Cf, bf16* __restrict__ Cb, int ldc, int ctrans,
               int M, int N, int Kpad, int act)
{
  int tiles_n = (N + 15) >> 4;
  int tiles_m = M >> 4;
  int tile = blockIdx.x * 8 + (threadIdx.x >> 5);
  if (tile >= tiles_m * tiles_n) return;
  int tm = (tile / tiles_n) << 4;
  int tn = (tile % tiles_n) << 4;
  int lane = threadIdx.x & 31;
  int half = lane >> 4;
  int l15  = lane & 15;

  const bf16* Ar = A  + (size_t)(tm + l15) * lda;   // A: M = lane&15 (both halves)
  int ncol = tn + l15;
  const bf16* Br = Bt + (size_t)ncol * ldb;         // B: N = lane&15 (both halves)

  v8f acc = {};
  for (int kk = 0; kk < Kpad; kk += 32) {
    int kb  = kk + half * 8;                        // A: K = kb..kb+7, kb+16..kb+23
    int kbb = kk + half * 8;                        // B: K = kbb..kbb+7, kbb+16..kbb+23
    v8bf a0 = *(const v8bf*)(Ar + kb);
    v8bf a1 = *(const v8bf*)(Ar + kb + 16);
    v8bf b0 = *(const v8bf*)(Br + kbb);
    v8bf b1 = *(const v8bf*)(Br + kbb + 16);
    v16bf a, b;
#pragma unroll
    for (int i = 0; i < 8; ++i) {
      a[i] = a0[i]; a[i + 8] = a1[i];
      b[i] = b0[i]; b[i + 8] = b1[i];
    }
    acc = __builtin_amdgcn_wmma_f32_16x16x32_bf16(false, a, false, b,
                                                  (short)0, acc, false, false);
  }

  if (ncol >= N) return;
  float bv = bias ? bias[ncol] : 0.0f;
  if (ctrans) {                                     // bf16, C^T, contiguous in m
    v8bf pk;
#pragma unroll
    for (int r = 0; r < 8; ++r) {
      float v = acc[r] + bv;
      if (act == 1) v = 1.0f / (1.0f + __expf(-v));
      else if (act == 2) v = tanhf(v);
      pk[r] = (bf16)v;
    }
    *(v8bf*)(Cb + (size_t)ncol * ldc + tm + half * 8) = pk;
  } else {
#pragma unroll
    for (int r = 0; r < 8; ++r) {
      int m = tm + half * 8 + r;
      float v = acc[r] + bv;
      if (act == 1) v = 1.0f / (1.0f + __expf(-v));
      else if (act == 2) v = tanhf(v);
      if (Cf) Cf[(size_t)m * ldc + ncol] = v;
      if (Cb) Cb[(size_t)m * ldc + ncol] = (bf16)v;
    }
  }
}

// ---------------------------------------------------------------------------
// Batched adj matmul: X[b] = adj[b] (64x64 bf16) * F[b] (64x100) using
// fT [112, 32768] (feature-major), result bf16 into XH[row, colofs..+99]
// (row stride XHW).
// ---------------------------------------------------------------------------
__global__ __launch_bounds__(256)
void wmma_adj_bmm(const bf16* __restrict__ adjb,   // [B,64,64] bf16
                  const bf16* __restrict__ fT,     // [112, 32768]
                  bf16* __restrict__ XH, int colofs)
{
  const int tiles_n = 7, tpb = 28;                 // 4 m-tiles x 7 n-tiles
  int tile = blockIdx.x * 8 + (threadIdx.x >> 5);
  if (tile >= BBATCH * tpb) return;
  int b  = tile / tpb;
  int r  = tile % tpb;
  int tm = (r / tiles_n) << 4;
  int tn = (r % tiles_n) << 4;
  int lane = threadIdx.x & 31;
  int half = lane >> 4, l15 = lane & 15;

  const bf16* Ar = adjb + (size_t)b * 64 * 64 + (size_t)(tm + l15) * 64;
  int ncol = tn + l15;
  const bf16* Br = fT + (size_t)ncol * ROWS + (size_t)b * 64;

  v8f acc = {};
#pragma unroll
  for (int kk = 0; kk < 64; kk += 32) {
    int kb  = kk + half * 8;
    int kbb = kk + half * 8;
    v8bf a0 = *(const v8bf*)(Ar + kb);
    v8bf a1 = *(const v8bf*)(Ar + kb + 16);
    v8bf b0 = *(const v8bf*)(Br + kbb);
    v8bf b1 = *(const v8bf*)(Br + kbb + 16);
    v16bf a, bv;
#pragma unroll
    for (int i = 0; i < 8; ++i) {
      a[i] = a0[i]; a[i + 8] = a1[i];
      bv[i] = b0[i]; bv[i + 8] = b1[i];
    }
    acc = __builtin_amdgcn_wmma_f32_16x16x32_bf16(false, a, false, bv,
                                                  (short)0, acc, false, false);
  }
  if (ncol >= DD) return;
#pragma unroll
  for (int r8 = 0; r8 < 8; ++r8) {
    int m = tm + half * 8 + r8;
    XH[(size_t)(b * 64 + m) * XHW + colofs + ncol] = (bf16)acc[r8];
  }
}

// --------------------------- prep / elementwise ---------------------------

__global__ void cvt_bf16(const float* __restrict__ s, bf16* __restrict__ d, int n) {
  int t = blockIdx.x * blockDim.x + threadIdx.x;
  if (t < n) d[t] = (bf16)s[t];
}

__global__ void fill_zero_bf16(bf16* __restrict__ p, int n) {
  int t = blockIdx.x * blockDim.x + threadIdx.x;
  if (t < n) p[t] = (bf16)0.0f;
}

// src: row-major [K,N] fp32 -> dst: [Npad16, Kpad] bf16 (transposed, padded)
__global__ void cvt_t_pad(const float* __restrict__ src, bf16* __restrict__ dst,
                          int K, int N, int Kpad, int total) {
  int t = blockIdx.x * blockDim.x + threadIdx.x;
  if (t >= total) return;
  int n = t / Kpad, k = t % Kpad;
  dst[t] = (n < N && k < K) ? (bf16)src[(size_t)k * N + n] : (bf16)0.0f;
}

// embedding[1:] -> embT [NOUTP, KP] bf16 (rows are already output columns)
__global__ void cvt_embT(const float* __restrict__ emb, bf16* __restrict__ dst) {
  int t = blockIdx.x * blockDim.x + threadIdx.x;
  if (t >= NOUTP * KP) return;
  int n = t >> 7, k = t & (KP - 1);
  dst[t] = (n < NOUT && k < DD) ? (bf16)emb[(size_t)(n + 1) * DD + k] : (bf16)0.0f;
}

// fin = embedding[item]: fp32 [ROWS,100], bf16 padded [ROWS,128],
// and the h slot (cols 200..299 + zero pad 300..319) of xh [ROWS,320].
__global__ void gather_fin(const float* __restrict__ emb, const int* __restrict__ item,
                           float* __restrict__ finf, bf16* __restrict__ finb,
                           bf16* __restrict__ xh) {
  int t = blockIdx.x * blockDim.x + threadIdx.x;
  if (t >= ROWS * KP) return;
  int r = t >> 7, d = t & (KP - 1);
  float v = 0.0f;
  if (d < DD) { v = emb[(size_t)item[r] * DD + d]; finf[(size_t)r * DD + d] = v; }
  finb[t] = (bf16)v;
  if (d < XHW - 200) xh[(size_t)r * XHW + 200 + d] = (bf16)v;   // zeros for d>=100
}

// xrh [ROWS,320]: cols<200 copy x, 200..299 = r*h, >=300 zero
__global__ void build_xrh(const bf16* __restrict__ xh, const float* __restrict__ ru,
                          const float* __restrict__ fin, bf16* __restrict__ xrh) {
  int t = blockIdx.x * blockDim.x + threadIdx.x;
  if (t >= ROWS * XHW) return;
  int r = t / XHW, c = t % XHW;
  if (c < 200)      xrh[t] = xh[t];
  else if (c < 300) {
    int d = c - 200;
    xrh[t] = (bf16)(ru[(size_t)r * 200 + d] * fin[(size_t)r * DD + d]);
  } else            xrh[t] = (bf16)0.0f;
}

__global__ void gru_update(const float* __restrict__ ru, const float* __restrict__ cf,
                           float* __restrict__ fin) {
  int t = blockIdx.x * blockDim.x + threadIdx.x;
  if (t >= ROWS * DD) return;
  int r = t / DD, d = t % DD;
  float u = ru[(size_t)r * 200 + 100 + d];
  fin[t] = u * fin[t] + (1.0f - u) * cf[t];
}

__global__ void compute_lastid(const float* __restrict__ mask, const int* __restrict__ alias,
                               int* __restrict__ lastid) {
  int b = blockIdx.x * blockDim.x + threadIdx.x;
  if (b >= BBATCH) return;
  float s = 0.0f;
  for (int l = 0; l < NN; ++l) s += mask[b * NN + l];
  int rm = (int)(s + 0.5f);
  lastid[b] = alias[b * NN + (rm - 1)];
}

__global__ void gather_seq(const float* __restrict__ fin, const int* __restrict__ alias,
                           float* __restrict__ seqf, bf16* __restrict__ seqb) {
  int t = blockIdx.x * blockDim.x + threadIdx.x;
  if (t >= ROWS * KP) return;
  int r = t >> 7, d = t & (KP - 1);
  float v = 0.0f;
  if (d < DD) {
    int b = r / NN, l = r % NN;
    int a = alias[b * NN + l];
    v = fin[((size_t)b * NN + a) * DD + d];
    seqf[(size_t)r * DD + d] = v;
  }
  seqb[t] = (bf16)v;
}

__global__ void gather_last(const float* __restrict__ fin, const int* __restrict__ lastid,
                            bf16* __restrict__ lastb) {
  int t = blockIdx.x * blockDim.x + threadIdx.x;
  if (t >= BBATCH * KP) return;
  int b = t >> 7, d = t & (KP - 1);
  lastb[t] = (d < DD) ? (bf16)fin[((size_t)b * NN + lastid[b]) * DD + d] : (bf16)0.0f;
}

__global__ void attn_coef(const float* __restrict__ last, const float* __restrict__ seq,
                          const float* __restrict__ nb, const float* __restrict__ nv,
                          const float* __restrict__ mask, float* __restrict__ coef) {
  int t = blockIdx.x * blockDim.x + threadIdx.x;
  if (t >= ROWS) return;
  int b = t / NN;
  float s = 0.0f;
  for (int d = 0; d < DD; ++d) {
    float m = 1.0f / (1.0f + __expf(-(last[(size_t)b * DD + d] +
                                      seq[(size_t)t * DD + d] + nb[d])));
    s += m * nv[d];
  }
  coef[t] = s * mask[t];
}

__global__ void compute_ma(const float* __restrict__ coef, const float* __restrict__ seqh,
                           bf16* __restrict__ mab) {
  int t = blockIdx.x * blockDim.x + threadIdx.x;
  if (t >= BBATCH * KP) return;
  int b = t >> 7, d = t & (KP - 1);
  float s = 0.0f;
  if (d < DD)
    for (int l = 0; l < NN; ++l)
      s += coef[b * NN + l] * seqh[((size_t)b * NN + l) * DD + d];
  mab[t] = (bf16)s;
}

// --------------------------- softmax / loss ---------------------------

__global__ __launch_bounds__(256)
void row_max(const float* __restrict__ logits, float* __restrict__ rowmax) {
  int b = blockIdx.x;
  __shared__ float sm[256];
  float m = -1e30f;
  for (int n = threadIdx.x; n < NOUT; n += 256)
    m = fmaxf(m, logits[(size_t)b * NOUT + n]);
  sm[threadIdx.x] = m; __syncthreads();
  for (int s = 128; s > 0; s >>= 1) {
    if (threadIdx.x < s) sm[threadIdx.x] = fmaxf(sm[threadIdx.x], sm[threadIdx.x + s]);
    __syncthreads();
  }
  if (threadIdx.x == 0) rowmax[b] = sm[0];
}

__global__ __launch_bounds__(256)
void row_sumexp(const float* __restrict__ logits, const float* __restrict__ rowmax,
                float* __restrict__ rowsum) {
  int b = blockIdx.x;
  __shared__ float sm[256];
  float mx = rowmax[b];
  float s = 0.0f;
  for (int n = threadIdx.x; n < NOUT; n += 256)
    s += __expf(logits[(size_t)b * NOUT + n] - mx);
  sm[threadIdx.x] = s; __syncthreads();
  for (int st = 128; st > 0; st >>= 1) {
    if (threadIdx.x < st) sm[threadIdx.x] += sm[threadIdx.x + st];
    __syncthreads();
  }
  if (threadIdx.x == 0) rowsum[b] = sm[0];
}

__global__ __launch_bounds__(256)
void loss_kernel(const float* __restrict__ logits, const int* __restrict__ tar,
                 const float* __restrict__ rowmax, const float* __restrict__ rowsum,
                 float* __restrict__ out0) {
  __shared__ float sm[256];
  float s = 0.0f;
  for (int b = threadIdx.x; b < BBATCH; b += 256) {
    int lb = tar[b] - 1;
    s -= logits[(size_t)b * NOUT + lb] - rowmax[b] - __logf(rowsum[b]);
  }
  sm[threadIdx.x] = s; __syncthreads();
  for (int st = 128; st > 0; st >>= 1) {
    if (threadIdx.x < st) sm[threadIdx.x] += sm[threadIdx.x + st];
    __syncthreads();
  }
  if (threadIdx.x == 0) out0[0] = sm[0] / (float)BBATCH;
}

// --------------------------- host launcher ---------------------------

static inline int ceil_div(long a, long b) { return (int)((a + b - 1) / b); }

extern "C" void kernel_launch(void* const* d_in, const int* in_sizes, int n_in,
                              void* d_out, int out_size, void* d_ws, size_t ws_size,
                              hipStream_t stream) {
  (void)in_sizes; (void)n_in; (void)out_size; (void)ws_size;
  const float* adj_in    = (const float*)d_in[0];
  const float* adj_out   = (const float*)d_in[1];
  const int*   item      = (const int*)  d_in[2];
  const int*   alias_    = (const int*)  d_in[3];
  const float* mask      = (const float*)d_in[4];
  const int*   tar       = (const int*)  d_in[5];
  const float* embedding = (const float*)d_in[6];
  const float* W_in      = (const float*)d_in[7];
  const float* b_in      = (const float*)d_in[8];
  const float* W_out     = (const float*)d_in[9];
  const float* b_out     = (const float*)d_in[10];
  const float* gate_k    = (const float*)d_in[11];
  const float* gate_b    = (const float*)d_in[12];
  const float* cand_k    = (const float*)d_in[13];
  const float* cand_b    = (const float*)d_in[14];
  const float* nasr_w1   = (const float*)d_in[15];
  const float* nasr_w2   = (const float*)d_in[16];
  const float* nasr_v    = (const float*)d_in[17];
  const float* nasr_b    = (const float*)d_in[18];

  float* out    = (float*)d_out;
  float* logits = out + 1;                       // [B, 49999]

  char*  w   = (char*)d_ws;
  size_t off = 0;
  auto alloc = [&](size_t bytes) -> void* {
    void* p = w + off;
    off += bytes;
    off = (off + 255) & ~(size_t)255;
    return p;
  };
  bf16*  embT   = (bf16*)alloc((size_t)NOUTP * KP * 2);     // [50000,128]
  bf16*  winT   = (bf16*)alloc((size_t)NFP * KP * 2);       // [112,128]
  bf16*  woutT  = (bf16*)alloc((size_t)NFP * KP * 2);
  bf16*  gkT    = (bf16*)alloc((size_t)208 * K3P * 2);      // [208,320]
  bf16*  ckT    = (bf16*)alloc((size_t)NFP * K3P * 2);      // [112,320]
  bf16*  w1T    = (bf16*)alloc((size_t)NFP * KP * 2);
  bf16*  w2T    = (bf16*)alloc((size_t)NFP * KP * 2);
  bf16*  adjinb = (bf16*)alloc((size_t)BBATCH * 64 * 64 * 2);
  bf16*  adjoutb= (bf16*)alloc((size_t)BBATCH * 64 * 64 * 2);
  float* fin_f  = (float*)alloc((size_t)ROWS * DD * 4);
  bf16*  finb   = (bf16*) alloc((size_t)ROWS * KP * 2);     // [32768,128]
  bf16*  xh     = (bf16*) alloc((size_t)ROWS * XHW * 2);    // [32768,320]
  bf16*  fiT    = (bf16*) alloc((size_t)NFP * ROWS * 2);    // [112,32768]
  bf16*  foT    = (bf16*) alloc((size_t)NFP * ROWS * 2);
  float* ru_f   = (float*)alloc((size_t)ROWS * 200 * 4);
  bf16*  xrh    = (bf16*) alloc((size_t)ROWS * XHW * 2);
  float* c_f    = (float*)alloc((size_t)ROWS * DD * 4);
  float* seqh_f = (float*)alloc((size_t)ROWS * DD * 4);
  bf16*  seqh_b = (bf16*) alloc((size_t)ROWS * KP * 2);
  bf16*  last_b = (bf16*) alloc((size_t)BBATCH * KP * 2);
  float* last_f = (float*)alloc((size_t)BBATCH * DD * 4);
  float* seq_f  = (float*)alloc((size_t)ROWS * DD * 4);
  float* coef_f = (float*)alloc((size_t)ROWS * 4);
  int*   lid_i  = (int*)  alloc(BBATCH * 4);
  bf16*  ma_b   = (bf16*) alloc((size_t)BBATCH * KP * 2);
  float* rmax_f = (float*)alloc(BBATCH * 4);
  float* rsum_f = (float*)alloc(BBATCH * 4);

  const int T = 256;
  auto gemm_grid = [](long M, long N) { return (int)(((M / 16) * ((N + 15) / 16) + 7) / 8); };

  // 1) operand prep (transpose + bf16 + zero padding)
  cvt_embT<<<ceil_div((long)NOUTP * KP, T), T, 0, stream>>>(embedding, embT);
  cvt_t_pad<<<ceil_div(NFP * KP, T), T, 0, stream>>>(W_in,   winT, DD, DD, KP, NFP * KP);
  cvt_t_pad<<<ceil_div(NFP * KP, T), T, 0, stream>>>(W_out,  woutT, DD, DD, KP, NFP * KP);
  cvt_t_pad<<<ceil_div(208 * K3P, T), T, 0, stream>>>(gate_k, gkT, K3, 200, K3P, 208 * K3P);
  cvt_t_pad<<<ceil_div(NFP * K3P, T), T, 0, stream>>>(cand_k, ckT, K3, DD, K3P, NFP * K3P);
  cvt_t_pad<<<ceil_div(NFP * KP, T), T, 0, stream>>>(nasr_w1, w1T, DD, DD, KP, NFP * KP);
  cvt_t_pad<<<ceil_div(NFP * KP, T), T, 0, stream>>>(nasr_w2, w2T, DD, DD, KP, NFP * KP);
  cvt_bf16<<<ceil_div((long)BBATCH * 64 * 64, T), T, 0, stream>>>(adj_in,  adjinb,  BBATCH * 64 * 64);
  cvt_bf16<<<ceil_div((long)BBATCH * 64 * 64, T), T, 0, stream>>>(adj_out, adjoutb, BBATCH * 64 * 64);
  fill_zero_bf16<<<ceil_div((long)(NFP - DD) * ROWS, T), T, 0, stream>>>(fiT + (size_t)DD * ROWS, (NFP - DD) * ROWS);
  fill_zero_bf16<<<ceil_div((long)(NFP - DD) * ROWS, T), T, 0, stream>>>(foT + (size_t)DD * ROWS, (NFP - DD) * ROWS);

  // 2) fin = embedding[item]
  gather_fin<<<ceil_div((long)ROWS * KP, T), T, 0, stream>>>(embedding, item, fin_f, finb, xh);

  // 3) fiT = (fin@W_in + b_in)^T ; foT = (fin@W_out + b_out)^T
  wmma_gemm<<<gemm_grid(ROWS, DD), T, 0, stream>>>(finb, KP, winT, KP, b_in,
                                                   nullptr, fiT, ROWS, 1, ROWS, DD, KP, 0);
  wmma_gemm<<<gemm_grid(ROWS, DD), T, 0, stream>>>(finb, KP, woutT, KP, b_out,
                                                   nullptr, foT, ROWS, 1, ROWS, DD, KP, 0);

  // 4) x = [adj_in@fi | adj_out@fo] into xh cols 0..199
  {
    int grid = (BBATCH * 28 + 7) / 8;
    wmma_adj_bmm<<<grid, T, 0, stream>>>(adjinb,  fiT, xh, 0);
    wmma_adj_bmm<<<grid, T, 0, stream>>>(adjoutb, foT, xh, DD);
  }

  // 5) gates = sigmoid([x|h] @ gate_kernel + gate_bias) -> ru_f [ROWS,200]
  wmma_gemm<<<gemm_grid(ROWS, 200), T, 0, stream>>>(xh, XHW, gkT, K3P, gate_b,
                                                    ru_f, nullptr, 200, 0, ROWS, 200, K3P, 1);

  // 6) xrh = [x | r*h | 0]
  build_xrh<<<ceil_div((long)ROWS * XHW, T), T, 0, stream>>>(xh, ru_f, fin_f, xrh);

  // 7) c = tanh(xrh @ cand_kernel + cand_bias)
  wmma_gemm<<<gemm_grid(ROWS, DD), T, 0, stream>>>(xrh, XHW, ckT, K3P, cand_b,
                                                   c_f, nullptr, DD, 0, ROWS, DD, K3P, 2);

  // 8) fin = u*h + (1-u)*c
  gru_update<<<ceil_div((long)ROWS * DD, T), T, 0, stream>>>(ru_f, c_f, fin_f);

  // 9) attention readout
  compute_lastid<<<ceil_div(BBATCH, T), T, 0, stream>>>(mask, alias_, lid_i);
  gather_seq<<<ceil_div((long)ROWS * KP, T), T, 0, stream>>>(fin_f, alias_, seqh_f, seqh_b);
  gather_last<<<ceil_div((long)BBATCH * KP, T), T, 0, stream>>>(fin_f, lid_i, last_b);

  wmma_gemm<<<gemm_grid(BBATCH, DD), T, 0, stream>>>(last_b, KP, w1T, KP, nullptr,
                                                     last_f, nullptr, DD, 0, BBATCH, DD, KP, 0);
  wmma_gemm<<<gemm_grid(ROWS, DD), T, 0, stream>>>(seqh_b, KP, w2T, KP, nullptr,
                                                   seq_f, nullptr, DD, 0, ROWS, DD, KP, 0);

  attn_coef<<<ceil_div(ROWS, T), T, 0, stream>>>(last_f, seq_f, nasr_b, nasr_v, mask, coef_f);
  compute_ma<<<ceil_div((long)BBATCH * KP, T), T, 0, stream>>>(coef_f, seqh_f, ma_b);

  // 10) logits = ma @ embedding[1:].T  (embT is already [N,K] layout)
  wmma_gemm<<<gemm_grid(BBATCH, NOUT), T, 0, stream>>>(ma_b, KP, embT, KP, nullptr,
                                                       logits, nullptr, NOUT, 0, BBATCH, NOUT, KP, 0);

  // 11) loss = -mean(log_softmax(logits)[rows, tar-1])
  row_max   <<<BBATCH, T, 0, stream>>>(logits, rmax_f);
  row_sumexp<<<BBATCH, T, 0, stream>>>(logits, rmax_f, rsum_f);
  loss_kernel<<<1, T, 0, stream>>>(logits, tar, rmax_f, rsum_f, out);
}
